// QFCModel_39470749450875
// MI455X (gfx1250) — matrix operand order repaired and checked by reading, not compiled, not run on hardware
//
#include <hip/hip_runtime.h>
#include <math.h>

// MI455X (gfx1250) — batched complex state evolution.
// ~206 GFLOP of GEMM vs ~11us of HBM traffic (U's L2-resident at 192MB)
// => compute bound => all GEMM flops through v_wmma_f32_16x16x32_f16,
// with double-buffered LDS fed by global_load_async_to_lds_b128 (ASYNCcnt).

#define DIM  4096
#define BSZ  1024
#define NW   12
#define BM   128   // batch tile (WMMA M)
#define BN   64    // output-amplitude tile (WMMA N)
#define BK   32    // K step (WMMA K)
#define NK   (DIM / BK)
#define AROW 40    // BK + 8 f16 LDS row padding (avoid bank conflicts)

typedef __attribute__((ext_vector_type(16))) _Float16 v16h;
typedef __attribute__((ext_vector_type(8)))  float    v8f;
typedef __attribute__((ext_vector_type(4)))  _Float16 h4;

union Frag { v16h v; uint4 q[2]; };

// ---- CDNA5 async global->LDS copy (16B per lane), tracked by ASYNCcnt ----
__device__ __forceinline__ void async_ld16(void* lds_dst, const void* gsrc) {
  uint32_t l = (uint32_t)(uintptr_t)(__attribute__((address_space(3))) const char*)lds_dst;
  asm volatile("global_load_async_to_lds_b128 %0, %1, off"
               :: "v"(l), "v"(gsrc) : "memory");
}
__device__ __forceinline__ void wait_async0() {
  asm volatile("s_wait_asynccnt 0" ::: "memory");
}

// Inverse of the composed CNOT-ring permutation: reference applies gates
// (c,t=(c+1)%NW) in reversed order; each CNOT is an involution, so the
// inverse applies them in forward order.
__device__ __forceinline__ int invperm(int i) {
#pragma unroll
  for (int c = 0; c < NW; ++c) {
    int t  = (c + 1) % NW;
    int cb = (i >> (NW - 1 - c)) & 1;
    i ^= cb << (NW - 1 - t);
  }
  return i;
}

__device__ __forceinline__ Frag neg_frag(const Frag& f) {
  Frag o;
#pragma unroll
  for (int k = 0; k < 2; ++k) {
    o.q[k].x = f.q[k].x ^ 0x80008000u;
    o.q[k].y = f.q[k].y ^ 0x80008000u;
    o.q[k].z = f.q[k].z ^ 0x80008000u;
    o.q[k].w = f.q[k].w ^ 0x80008000u;
  }
  return o;
}

__device__ __forceinline__ void cvt_store8(_Float16* dst, float4 a, float4 b) {
  h4 lo, hi;
  lo.x = (_Float16)a.x; lo.y = (_Float16)a.y; lo.z = (_Float16)a.z; lo.w = (_Float16)a.w;
  hi.x = (_Float16)b.x; hi.y = (_Float16)b.y; hi.z = (_Float16)b.z; hi.w = (_Float16)b.w;
  *(h4*)dst       = lo;
  *(h4*)(dst + 4) = hi;
}

// ---------------- normalize: s0[b,:] = x[b,:] / ||x[b,:]|| (f16) -----------
__global__ __launch_bounds__(256) void normalize_k(const float* __restrict__ x,
                                                   _Float16* __restrict__ s0) {
  const int b   = blockIdx.x;
  const int tid = threadIdx.x;
  const float4* xr = (const float4*)(x + (size_t)b * DIM);
  float4 v[4];
  float ss = 0.f;
#pragma unroll
  for (int it = 0; it < 4; ++it) {
    v[it] = xr[tid + it * 256];
    ss += v[it].x * v[it].x + v[it].y * v[it].y + v[it].z * v[it].z + v[it].w * v[it].w;
  }
#pragma unroll
  for (int off = 16; off > 0; off >>= 1) ss += __shfl_xor(ss, off, 32);
  __shared__ float red[8];
  if ((tid & 31) == 0) red[tid >> 5] = ss;
  __syncthreads();
  float tot = 0.f;
#pragma unroll
  for (int w = 0; w < 8; ++w) tot += red[w];
  const float rinv = rsqrtf(tot);
  h4* so = (h4*)(s0 + (size_t)b * DIM);
#pragma unroll
  for (int it = 0; it < 4; ++it) {
    h4 o;
    o.x = (_Float16)(v[it].x * rinv);
    o.y = (_Float16)(v[it].y * rinv);
    o.z = (_Float16)(v[it].z * rinv);
    o.w = (_Float16)(v[it].w * rinv);
    so[tid + it * 256] = o;
  }
}

// ---------------- GEMM1: p = perm(U @ s) for U_re and U_im -----------------
// C[m=batch, n=amp] = sum_k S[m,k] * U[n,k]   (A = state rows, B = U rows)
// Double-buffered: A tile via async global->LDS; B tile (f32->f16) staged in
// registers for k+1 while tile k computes. One barrier per K step.
__global__ __launch_bounds__(256) void gemm1_k(const _Float16* __restrict__ s0,
                                               const float* __restrict__ Ure,
                                               const float* __restrict__ Uim,
                                               _Float16* __restrict__ pre,
                                               _Float16* __restrict__ pim) {
  __shared__ _Float16 lsA[2][BM * AROW];       // state tile ping/pong
  __shared__ _Float16 lsB[2][2 * BN * AROW];   // U_re / U_im tiles ping/pong

  const int tid  = threadIdx.x;
  const int lane = tid & 31;
  const int wid  = tid >> 5;
  const int wm   = wid & 3;   // 4 waves over M
  const int wn   = wid >> 2;  // 2 waves over N
  const int bm0  = blockIdx.y * BM;
  const int n0   = blockIdx.x * BN;
  const int fr   = lane & 15, fh = lane >> 4;

  const int arow0 = (tid >> 2), aoff = (tid & 3);            // A: 2 chunks/thread
  const int brow  = (tid >> 2), boff = (tid & 3);            // B: 1 chunk/thread/mat
  const size_t bgoff = (size_t)(n0 + brow) * DIM + boff * 8;

  v8f accre[2][2], accim[2][2];
#pragma unroll
  for (int i = 0; i < 2; ++i)
#pragma unroll
    for (int j = 0; j < 2; ++j) {
#pragma unroll
      for (int r = 0; r < 8; ++r) { accre[i][j][r] = 0.f; accim[i][j][r] = 0.f; }
    }

  float4 breg[4];
  // ---- prologue: fill buffer 0 for k0 = 0 ----
#pragma unroll
  for (int it = 0; it < 2; ++it) {
    int row = arow0 + it * 64;
    async_ld16(&lsA[0][row * AROW + aoff * 8],
               s0 + (size_t)(bm0 + row) * DIM + aoff * 8);
  }
  {
    const float4* pr = (const float4*)(Ure + bgoff);
    const float4* pi = (const float4*)(Uim + bgoff);
    breg[0] = pr[0]; breg[1] = pr[1]; breg[2] = pi[0]; breg[3] = pi[1];
    cvt_store8(&lsB[0][brow * AROW + boff * 8], breg[0], breg[1]);
    cvt_store8(&lsB[0][BN * AROW + brow * AROW + boff * 8], breg[2], breg[3]);
  }
  wait_async0();
  __syncthreads();

  for (int kk = 0; kk < NK; ++kk) {
    const int cur = kk & 1, nxt = cur ^ 1;
    const int k1  = (kk + 1) * BK;
    const bool more = (kk + 1 < NK);
    if (more) {
      // prefetch next A tile straight into LDS (ASYNCcnt path)
#pragma unroll
      for (int it = 0; it < 2; ++it) {
        int row = arow0 + it * 64;
        async_ld16(&lsA[nxt][row * AROW + aoff * 8],
                   s0 + (size_t)(bm0 + row) * DIM + k1 + aoff * 8);
      }
      // issue next U (f32) loads; converted/stored after compute
      const float4* pr = (const float4*)(Ure + bgoff + k1);
      const float4* pi = (const float4*)(Uim + bgoff + k1);
      breg[0] = pr[0]; breg[1] = pr[1]; breg[2] = pi[0]; breg[3] = pi[1];
    }

    // ---- compute on current buffer ----
    Frag af[2], bre[2], bim[2];
#pragma unroll
    for (int i = 0; i < 2; ++i) {
      int base = (wm * 32 + i * 16 + fr) * AROW;
      af[i].q[0] = *(const uint4*)&lsA[cur][base + fh * 8];       // K = h*8..
      af[i].q[1] = *(const uint4*)&lsA[cur][base + 16 + fh * 8];  // K = 16+h*8..
    }
#pragma unroll
    for (int j = 0; j < 2; ++j) {
      int nb = (wn * 32 + j * 16 + fr) * AROW + fh * 8;           // K = h*8.., 16+h*8..
      bre[j].q[0] = *(const uint4*)&lsB[cur][nb];
      bre[j].q[1] = *(const uint4*)&lsB[cur][nb + 16];
      bim[j].q[0] = *(const uint4*)&lsB[cur][BN * AROW + nb];
      bim[j].q[1] = *(const uint4*)&lsB[cur][BN * AROW + nb + 16];
    }
#pragma unroll
    for (int i = 0; i < 2; ++i)
#pragma unroll
      for (int j = 0; j < 2; ++j) {
        accre[i][j] = __builtin_amdgcn_wmma_f32_16x16x32_f16(
            false, af[i].v, false, bre[j].v, (short)0, accre[i][j], false, false);
        accim[i][j] = __builtin_amdgcn_wmma_f32_16x16x32_f16(
            false, af[i].v, false, bim[j].v, (short)0, accim[i][j], false, false);
      }

    if (more) {
      cvt_store8(&lsB[nxt][brow * AROW + boff * 8], breg[0], breg[1]);
      cvt_store8(&lsB[nxt][BN * AROW + brow * AROW + boff * 8], breg[2], breg[3]);
      wait_async0();
    }
    __syncthreads();
  }

  // Epilogue: f16 store with the amplitude permutation folded in (scatter cols)
#pragma unroll
  for (int i = 0; i < 2; ++i)
#pragma unroll
    for (int j = 0; j < 2; ++j) {
      int n  = n0 + wn * 32 + j * 16 + fr;
      int pn = invperm(n);
      int mb = bm0 + wm * 32 + i * 16 + fh * 8;
#pragma unroll
      for (int r = 0; r < 8; ++r) {
        size_t o = (size_t)(mb + r) * DIM + pn;
        pre[o] = (_Float16)accre[i][j][r];
        pim[o] = (_Float16)accim[i][j][r];
      }
    }
}

// ---------------- GEMM2: out = |U1 @ p| (full complex) ---------------------
// re += Are*Bre - Aim*Bim ; im += Are*Bim + Aim*Bre (Aim negated via sign xor)
__global__ __launch_bounds__(256) void gemm2_k(const _Float16* __restrict__ pre,
                                               const _Float16* __restrict__ pim,
                                               const float* __restrict__ U1re,
                                               const float* __restrict__ U1im,
                                               float* __restrict__ out) {
  __shared__ _Float16 lsA[2][2 * BM * AROW];   // p_re / p_im tiles ping/pong
  __shared__ _Float16 lsB[2][2 * BN * AROW];   // U1_re / U1_im tiles ping/pong

  const int tid  = threadIdx.x;
  const int lane = tid & 31;
  const int wid  = tid >> 5;
  const int wm   = wid & 3;
  const int wn   = wid >> 2;
  const int bm0  = blockIdx.y * BM;
  const int n0   = blockIdx.x * BN;
  const int fr   = lane & 15, fh = lane >> 4;

  const int arow0 = (tid >> 2), aoff = (tid & 3);
  const int brow  = (tid >> 2), boff = (tid & 3);
  const size_t bgoff = (size_t)(n0 + brow) * DIM + boff * 8;

  v8f accre[2][2], accim[2][2];
#pragma unroll
  for (int i = 0; i < 2; ++i)
#pragma unroll
    for (int j = 0; j < 2; ++j) {
#pragma unroll
      for (int r = 0; r < 8; ++r) { accre[i][j][r] = 0.f; accim[i][j][r] = 0.f; }
    }

  float4 breg[4];
  // ---- prologue: fill buffer 0 for k0 = 0 ----
#pragma unroll
  for (int it = 0; it < 2; ++it) {
    int row = arow0 + it * 64;
    size_t g = (size_t)(bm0 + row) * DIM + aoff * 8;
    async_ld16(&lsA[0][row * AROW + aoff * 8], pre + g);
    async_ld16(&lsA[0][BM * AROW + row * AROW + aoff * 8], pim + g);
  }
  {
    const float4* pr = (const float4*)(U1re + bgoff);
    const float4* pi = (const float4*)(U1im + bgoff);
    breg[0] = pr[0]; breg[1] = pr[1]; breg[2] = pi[0]; breg[3] = pi[1];
    cvt_store8(&lsB[0][brow * AROW + boff * 8], breg[0], breg[1]);
    cvt_store8(&lsB[0][BN * AROW + brow * AROW + boff * 8], breg[2], breg[3]);
  }
  wait_async0();
  __syncthreads();

  for (int kk = 0; kk < NK; ++kk) {
    const int cur = kk & 1, nxt = cur ^ 1;
    const int k1  = (kk + 1) * BK;
    const bool more = (kk + 1 < NK);
    if (more) {
#pragma unroll
      for (int it = 0; it < 2; ++it) {
        int row = arow0 + it * 64;
        size_t g = (size_t)(bm0 + row) * DIM + k1 + aoff * 8;
        async_ld16(&lsA[nxt][row * AROW + aoff * 8], pre + g);
        async_ld16(&lsA[nxt][BM * AROW + row * AROW + aoff * 8], pim + g);
      }
      const float4* pr = (const float4*)(U1re + bgoff + k1);
      const float4* pi = (const float4*)(U1im + bgoff + k1);
      breg[0] = pr[0]; breg[1] = pr[1]; breg[2] = pi[0]; breg[3] = pi[1];
    }

    // ---- compute on current buffer ----
    Frag are[2], aim[2], aimn[2], bre[2], bim[2];
#pragma unroll
    for (int i = 0; i < 2; ++i) {
      int base = (wm * 32 + i * 16 + fr) * AROW;
      are[i].q[0] = *(const uint4*)&lsA[cur][base + fh * 8];
      are[i].q[1] = *(const uint4*)&lsA[cur][base + 16 + fh * 8];
      aim[i].q[0] = *(const uint4*)&lsA[cur][BM * AROW + base + fh * 8];
      aim[i].q[1] = *(const uint4*)&lsA[cur][BM * AROW + base + 16 + fh * 8];
      aimn[i] = neg_frag(aim[i]);
    }
#pragma unroll
    for (int j = 0; j < 2; ++j) {
      int nb = (wn * 32 + j * 16 + fr) * AROW + fh * 8;
      bre[j].q[0] = *(const uint4*)&lsB[cur][nb];
      bre[j].q[1] = *(const uint4*)&lsB[cur][nb + 16];
      bim[j].q[0] = *(const uint4*)&lsB[cur][BN * AROW + nb];
      bim[j].q[1] = *(const uint4*)&lsB[cur][BN * AROW + nb + 16];
    }
#pragma unroll
    for (int i = 0; i < 2; ++i)
#pragma unroll
      for (int j = 0; j < 2; ++j) {
        accre[i][j] = __builtin_amdgcn_wmma_f32_16x16x32_f16(
            false, are[i].v, false, bre[j].v, (short)0, accre[i][j], false, false);
        accre[i][j] = __builtin_amdgcn_wmma_f32_16x16x32_f16(
            false, aimn[i].v, false, bim[j].v, (short)0, accre[i][j], false, false);
        accim[i][j] = __builtin_amdgcn_wmma_f32_16x16x32_f16(
            false, are[i].v, false, bim[j].v, (short)0, accim[i][j], false, false);
        accim[i][j] = __builtin_amdgcn_wmma_f32_16x16x32_f16(
            false, aim[i].v, false, bre[j].v, (short)0, accim[i][j], false, false);
      }

    if (more) {
      cvt_store8(&lsB[nxt][brow * AROW + boff * 8], breg[0], breg[1]);
      cvt_store8(&lsB[nxt][BN * AROW + brow * AROW + boff * 8], breg[2], breg[3]);
      wait_async0();
    }
    __syncthreads();
  }

  // Epilogue: magnitude, f32 coalesced stores
#pragma unroll
  for (int i = 0; i < 2; ++i)
#pragma unroll
    for (int j = 0; j < 2; ++j) {
      int n  = n0 + wn * 32 + j * 16 + fr;
      int mb = bm0 + wm * 32 + i * 16 + fh * 8;
#pragma unroll
      for (int r = 0; r < 8; ++r) {
        float re = accre[i][j][r], im = accim[i][j][r];
        out[(size_t)(mb + r) * DIM + n] = sqrtf(re * re + im * im);
      }
    }
}

// ---------------- host ------------------------------------------------------
extern "C" void kernel_launch(void* const* d_in, const int* in_sizes, int n_in,
                              void* d_out, int out_size, void* d_ws, size_t ws_size,
                              hipStream_t stream) {
  const float* x    = (const float*)d_in[0];
  const float* Ure  = (const float*)d_in[1];
  const float* Uim  = (const float*)d_in[2];
  const float* U1re = (const float*)d_in[3];
  const float* U1im = (const float*)d_in[4];
  float* out = (float*)d_out;

  // workspace: 3 x (1024*4096) f16 = 24 MB
  _Float16* s0  = (_Float16*)d_ws;
  _Float16* pre = s0  + (size_t)BSZ * DIM;
  _Float16* pim = pre + (size_t)BSZ * DIM;

  normalize_k<<<BSZ, 256, 0, stream>>>(x, s0);
  dim3 grid(DIM / BN, BSZ / BM);  // 64 x 8
  gemm1_k<<<grid, 256, 0, stream>>>(s0, Ure, Uim, pre, pim);
  gemm2_k<<<grid, 256, 0, stream>>>(pre, pim, U1re, U1im, out);
}
